// CarPlanner_61340722921625
// MI455X (gfx1250) — compile-verified, matrix operand order checked by reading, not run on hardware
//
#include <hip/hip_runtime.h>
#include <hip/hip_bf16.h>
#include <math.h>

// ---------------- problem constants (match reference) ----------------
#define DD        256
#define D_LANE    128
#define N_AG      64
#define K_NN      32
#define T_FUT     80
#define N_LANES   64
#define N_PTS     20
#define NHEAD     4
#define BB        128
#define NKEYS     (K_NN + N_LANES)   // 96

typedef __attribute__((ext_vector_type(16))) _Float16 v16h;
typedef __attribute__((ext_vector_type(8)))  _Float16 v8h;
typedef __attribute__((ext_vector_type(4)))  _Float16 v4h;
typedef __attribute__((ext_vector_type(8)))  float    v8f;

#if defined(__gfx1250__) && __has_builtin(__builtin_amdgcn_global_load_async_to_lds_b32)
#define HAVE_ASYNC_LDS 1
typedef __attribute__((address_space(1))) int* glob_i32p;
typedef __attribute__((address_space(3))) int* lds_i32p;
#else
#define HAVE_ASYNC_LDS 0
#endif

__device__ __forceinline__ float gelu_exact(float x) {
  return 0.5f * x * (1.0f + erff(x * 0.70710678118654752f));
}

__device__ __forceinline__ float wrapf(float h) {
  const float PI  = 3.14159265358979323846f;
  const float TAU = 6.28318530717958647692f;
  float w = fmodf(h + PI, TAU);
  if (w < 0.0f) w += TAU;
  return w - PI;
}

// =====================================================================
// WMMA GEMM:  Y = act(X @ W + bias) [* rowscale]
//   X: MxK row-major f32, W: KxN row-major f32 (din x dout), Y: MxN f32
//   REQUIRES: M % 128 == 0, N % 64 == 0, K % 32 == 0 (true for all calls)
//   Block: 256 threads (8 waves). Block tile 128x64, K-step 32.
//   Wave w owns rows [16w,16w+16) x all 64 cols -> 4 accumulators,
//   4 x v_wmma_f32_16x16x32_f16 per K-step from a single A fragment.
//   A staged coalesced f32->f16 into LDS; B staged transposed so each
//   lane's fragment is two aligned ds_load_b128.
//   act: 0=none 1=relu 2=gelu(exact)
// =====================================================================
__global__ __launch_bounds__(256)
void wmma_gemm(const float* __restrict__ X, const float* __restrict__ W,
               const float* __restrict__ bias, const float* __restrict__ rowscale,
               float* __restrict__ Y, int M, int N, int K, int act)
{
  __shared__ _Float16 sA[128][40];   // [row][k], 80B pitch (16B aligned)
  __shared__ _Float16 sB[64][40];    // transposed: [col][k]

  const int t    = threadIdx.x;
  const int lane = t & 31;
  const int wave = t >> 5;                 // 0..7
  const unsigned rowB = blockIdx.y * 128u; // block row base
  const unsigned colB = blockIdx.x * 64u;  // block col base

  // staging assignments
  const int ar = t >> 1;                   // A: row 0..127
  const int ak = (t & 1) * 16;             // A: k sub-block 0 / 16
  const int bk = t >> 4;                   // B: k row 0..15 (+16 on 2nd pass)
  const int bc = (t & 15) * 4;             // B: col group (4 cols)

  v8f acc0 = {}, acc1 = {}, acc2 = {}, acc3 = {};

  for (int k0 = 0; k0 < K; k0 += 32) {
    // ---- stage A tile (128x32), 16 contiguous floats per thread ----
    {
      const float* src = X + (unsigned)((rowB + ar) * (unsigned)K + k0 + ak);
#pragma unroll
      for (int i = 0; i < 16; i += 4) {
        float4 v = *(const float4*)(src + i);
        v4h h; h[0] = (_Float16)v.x; h[1] = (_Float16)v.y;
               h[2] = (_Float16)v.z; h[3] = (_Float16)v.w;
        *(v4h*)&sA[ar][ak + i] = h;        // ds_store_b64
      }
      if (k0 + 32 < K) __builtin_prefetch(src + 32, 0, 0);  // global_prefetch_b8
    }
    // ---- stage B tile (32x64) transposed, coalesced float4 per thread ----
#pragma unroll
    for (int it = 0; it < 2; ++it) {
      int kk = bk + it * 16;
      const float* src = W + (unsigned)((k0 + kk) * (unsigned)N + colB + bc);
      float4 v = *(const float4*)src;
      sB[bc + 0][kk] = (_Float16)v.x;
      sB[bc + 1][kk] = (_Float16)v.y;
      sB[bc + 2][kk] = (_Float16)v.z;
      sB[bc + 3][kk] = (_Float16)v.w;
      if (k0 + 32 < K) __builtin_prefetch(src + (size_t)32 * N, 0, 0);
    }
    __syncthreads();

    // ---- fragments + 4 WMMAs ----
    const int khi = (lane >> 4) * 8;       // K sub-offset per half-wave
    const int arow = (wave << 4) + (lane & 15);
    v8h alo = *(const v8h*)&sA[arow][khi];        // ds_load_b128
    v8h ahi = *(const v8h*)&sA[arow][khi + 16];   // ds_load_b128
    v16h a = __builtin_shufflevector(alo, ahi, 0,1,2,3,4,5,6,7,8,9,10,11,12,13,14,15);

#pragma unroll
    for (int g = 0; g < 4; ++g) {
      const int bcol = (g << 4) + (lane & 15);
      v8h blo = *(const v8h*)&sB[bcol][khi];
      v8h bhi = *(const v8h*)&sB[bcol][khi + 16];
      v16h b = __builtin_shufflevector(blo, bhi, 0,1,2,3,4,5,6,7,8,9,10,11,12,13,14,15);
      v8f& acc = (g == 0) ? acc0 : (g == 1) ? acc1 : (g == 2) ? acc2 : acc3;
      acc = __builtin_amdgcn_wmma_f32_16x16x32_f16(
          false, a, false, b, (short)0, acc, false, false);
    }
    __syncthreads();
  }

  // ---- epilogue: acc[r] -> row rowB+16*wave+8*(lane>>4)+r, col colB+16g+(lane&15)
  const unsigned rbase = rowB + (wave << 4) + ((lane >> 4) << 3);
#pragma unroll
  for (int g = 0; g < 4; ++g) {
    const v8f& acc = (g == 0) ? acc0 : (g == 1) ? acc1 : (g == 2) ? acc2 : acc3;
    const unsigned c = colB + (g << 4) + (lane & 15);
    const float bi = bias ? bias[c] : 0.0f;
#pragma unroll
    for (int r = 0; r < 8; ++r) {
      unsigned rr = rbase + r;
      float v = acc[r] + bi;
      if (act == 1)      v = fmaxf(v, 0.0f);
      else if (act == 2) v = gelu_exact(v);
      if (rowscale)      v *= rowscale[rr];
      Y[(size_t)rr * N + c] = v;
    }
  }
}

// =====================================================================
// Lane PointNet (hoisted; K=3 too small for WMMA): one block per (b,lane),
// 20 points through 3->32(relu)->64(relu)->128, maxpool, mask.
// =====================================================================
__global__ __launch_bounds__(128)
void lane_encoder(const float* __restrict__ lanes, const float* __restrict__ lmask,
                  const float* __restrict__ w1, const float* __restrict__ b1,
                  const float* __restrict__ w2, const float* __restrict__ b2,
                  const float* __restrict__ w3, const float* __restrict__ b3,
                  float* __restrict__ lane_feat)
{
  const int bl = blockIdx.x;                       // 0 .. B*N_LANES-1
  const float* P = lanes + (size_t)bl * N_PTS * 3;
  __shared__ float sp[3];
  __shared__ float sh1[32];
  __shared__ float sh2[64];
  const int t = threadIdx.x;                       // 0..127
  float best = -INFINITY;
  for (int p = 0; p < N_PTS; ++p) {
    if (t < 3) sp[t] = P[p * 3 + t];
    __syncthreads();
    if (t < 32) {
      float s = b1[t];
#pragma unroll
      for (int i = 0; i < 3; ++i) s += sp[i] * w1[i * 32 + t];
      sh1[t] = fmaxf(s, 0.0f);
    }
    __syncthreads();
    if (t < 64) {
      float s = b2[t];
      for (int i = 0; i < 32; ++i) s += sh1[i] * w2[i * 64 + t];
      sh2[t] = fmaxf(s, 0.0f);
    }
    __syncthreads();
    {
      float s = b3[t];
      for (int i = 0; i < 64; ++i) s += sh2[i] * w3[i * 128 + t];
      best = fmaxf(best, s);                       // last layer: no relu
    }
    __syncthreads();
  }
  lane_feat[(size_t)bl * D_LANE + t] = best * lmask[bl];
}

// =====================================================================
// Carry init: mq = mode_embed[mode_c], ego = 0
// =====================================================================
__global__ __launch_bounds__(256)
void init_state(const float* __restrict__ mode_embed, const int* __restrict__ mode_c,
                float* __restrict__ mq, float* __restrict__ ego)
{
  int i = blockIdx.x * blockDim.x + threadIdx.x;
  if (i < BB * DD) {
    int b = i / DD, d = i % DD;
    mq[i] = mode_embed[(size_t)mode_c[b] * DD + d];
  }
  if (i < BB * 3) ego[i] = 0.0f;
}

// =====================================================================
// Per-step: ego-frame transform + agent MLP layer 1 (K=4, scalar) + dist
// =====================================================================
__global__ __launch_bounds__(256)
void agent_pre(const float* __restrict__ agents_seq, int t,
               const float* __restrict__ ego,
               const float* __restrict__ w1, const float* __restrict__ b1,
               float* __restrict__ h1, float* __restrict__ dist)
{
  int i = blockIdx.x * blockDim.x + threadIdx.x;   // 0 .. B*N_AG-1
  if (i >= BB * N_AG) return;
  int b = i >> 6;
  const float* a = agents_seq + (((size_t)b * T_FUT + t) * N_AG + (i & 63)) * 4;
  float ex = ego[b * 3 + 0], ey = ego[b * 3 + 1], eh = ego[b * 3 + 2];
  float dx = a[0] - ex, dy = a[1] - ey;
  float c = cosf(-eh), s = sinf(-eh);
  float xe = c * dx - s * dy;
  float ye = s * dx + c * dy;
  float he = wrapf(a[2] - eh);
  float w  = a[3];
  dist[i] = sqrtf(xe * xe + ye * ye);
  float* hr = h1 + (size_t)i * 64;
  for (int o = 0; o < 64; ++o) {
    float v = b1[o] + xe * w1[0 * 64 + o] + ye * w1[1 * 64 + o]
                    + he * w1[2 * 64 + o] + w  * w1[3 * 64 + o];
    hr[o] = fmaxf(v, 0.0f);
  }
}

// =====================================================================
// Exact top-K=32 smallest dist per batch via rank counting (matches
// jax.lax.top_k(-dist) tie ordering), gather selected agent features.
// =====================================================================
__global__ __launch_bounds__(64)
void topk_gather(const float* __restrict__ dist, const float* __restrict__ pa,
                 float* __restrict__ pa_sel)
{
  const int b = blockIdx.x;
  const int n = threadIdx.x;
  __shared__ float d[N_AG];
  d[n] = dist[b * N_AG + n];
  __syncthreads();
  const float dn = d[n];
  int rank = 0;
  for (int j = 0; j < N_AG; ++j)
    rank += (d[j] < dn) || (d[j] == dn && j < n);
  if (rank < K_NN) {
    const float4* src = (const float4*)(pa + ((size_t)b * N_AG + n) * DD);
    float4* dst = (float4*)(pa_sel + ((size_t)b * K_NN + rank) * DD);
    for (int c = 0; c < DD / 4; ++c) dst[c] = src[c];
  }
}

// =====================================================================
// Cross-attention (4 heads, 96 keys): one block per batch element.
// Query staged into LDS via CDNA5 async global->LDS copy when available.
// =====================================================================
__global__ __launch_bounds__(256)
void attention(const float* __restrict__ q, const float* __restrict__ k_sel,
               const float* __restrict__ k_map, const float* __restrict__ v_sel,
               const float* __restrict__ v_map, float* __restrict__ attn_out)
{
  const int b = blockIdx.x;
  const int t = threadIdx.x;
  __shared__ float sq[DD];
  __shared__ float satt[NHEAD][NKEYS];
#if HAVE_ASYNC_LDS
  {
    float* gp = (float*)(q + (size_t)b * DD + t);
    __builtin_amdgcn_global_load_async_to_lds_b32(
        (glob_i32p)(void*)gp, (lds_i32p)(void*)&sq[t], 0, 0);
    asm volatile("s_wait_asynccnt 0x0" ::: "memory");
  }
#else
  sq[t] = q[(size_t)b * DD + t];
#endif
  __syncthreads();
  for (int j = t; j < NHEAD * NKEYS; j += 256) {
    int h = j / NKEYS, k = j % NKEYS;
    const float* kr = (k < K_NN) ? (k_sel + ((size_t)b * K_NN + k) * DD)
                                 : (k_map + ((size_t)b * N_LANES + (k - K_NN)) * DD);
    float s = 0.0f;
    for (int dd2 = 0; dd2 < 64; ++dd2) s += sq[h * 64 + dd2] * kr[h * 64 + dd2];
    satt[h][k] = s * 0.125f;                       // 1/sqrt(64)
  }
  __syncthreads();
  if (t < NHEAD) {
    float mx = -INFINITY;
    for (int k = 0; k < NKEYS; ++k) mx = fmaxf(mx, satt[t][k]);
    float sum = 0.0f;
    for (int k = 0; k < NKEYS; ++k) { float e = expf(satt[t][k] - mx); satt[t][k] = e; sum += e; }
    float inv = 1.0f / sum;
    for (int k = 0; k < NKEYS; ++k) satt[t][k] *= inv;
  }
  __syncthreads();
  {
    const int h = t >> 6;
    float o = 0.0f;
    for (int k = 0; k < NKEYS; ++k) {
      const float* vr = (k < K_NN) ? (v_sel + ((size_t)b * K_NN + k) * DD)
                                   : (v_map + ((size_t)b * N_LANES + (k - K_NN)) * DD);
      o += satt[h][k] * vr[t];
    }
    attn_out[(size_t)b * DD + t] = o;
  }
}

// =====================================================================
// y = LayerNorm(x + r) * g + b  (one block per batch element, D=256)
// =====================================================================
__global__ __launch_bounds__(256)
void add_ln(const float* __restrict__ x, const float* __restrict__ r,
            const float* __restrict__ g, const float* __restrict__ bb,
            float* __restrict__ y)
{
  const int b = blockIdx.x;
  const int t = threadIdx.x;
  __shared__ float sv[DD];
  __shared__ float stats[2];
  float v = x[(size_t)b * DD + t] + r[(size_t)b * DD + t];
  sv[t] = v;
  __syncthreads();
  if (t == 0) {
    float m = 0.0f;
    for (int i = 0; i < DD; ++i) m += sv[i];
    m *= (1.0f / DD);
    float var = 0.0f;
    for (int i = 0; i < DD; ++i) { float d0 = sv[i] - m; var += d0 * d0; }
    var *= (1.0f / DD);
    stats[0] = m;
    stats[1] = 1.0f / sqrtf(var + 1e-5f);
  }
  __syncthreads();
  y[(size_t)b * DD + t] = (v - stats[0]) * stats[1] * g[t] + bb[t];
}

// =====================================================================
// Action head tail (256->3) + pose integration + trajectory write.
// =====================================================================
__global__ __launch_bounds__(32)
void action_final(const float* __restrict__ act_h, const float* __restrict__ w2,
                  const float* __restrict__ b2, float* __restrict__ ego,
                  float* __restrict__ out, int t)
{
  const int b = blockIdx.x;
  const int l = threadIdx.x;
  __shared__ float a[3];
  if (l < 3) {
    float s = b2[l];
    const float* hr = act_h + (size_t)b * DD;
    for (int i = 0; i < DD; ++i) s += hr[i] * w2[i * 3 + l];
    a[l] = s;
  }
  __syncthreads();
  if (l == 0) {
    float ex = ego[b * 3 + 0], ey = ego[b * 3 + 1], eh = ego[b * 3 + 2];
    float cb = cosf(eh), sb = sinf(eh);
    float xg = cb * a[0] - sb * a[1] + ex;
    float yg = sb * a[0] + cb * a[1] + ey;
    float hg = wrapf(a[2] + eh);
    float* o = out + ((size_t)b * T_FUT + t) * 3;
    o[0] = xg; o[1] = yg; o[2] = hg;
    ego[b * 3 + 0] = xg; ego[b * 3 + 1] = yg; ego[b * 3 + 2] = hg;
  }
}

// =====================================================================
// Host orchestration
// =====================================================================
extern "C" void kernel_launch(void* const* d_in, const int* in_sizes, int n_in,
                              void* d_out, int out_size, void* d_ws, size_t ws_size,
                              hipStream_t stream) {
  (void)in_sizes; (void)n_in; (void)out_size; (void)ws_size;

  // ---- inputs (setup_inputs dict order, params in insertion order) ----
  const float* agents_seq = (const float*)d_in[0];   // (B,T,N,4)
  const float* agents_mask= (const float*)d_in[1];   // (B,N)
  const int*   mode_c     = (const int*)  d_in[2];   // (B,)
  const float* map_lanes  = (const float*)d_in[3];   // (B,NL,NP,3)
  const float* lanes_mask = (const float*)d_in[4];   // (B,NL)
  const float* mode_embed = (const float*)d_in[5];   // (6,256)
  const float* aw1=(const float*)d_in[6],  *ab1=(const float*)d_in[7];   // 4x64
  const float* aw2=(const float*)d_in[8],  *ab2=(const float*)d_in[9];   // 64x128
  const float* aw3=(const float*)d_in[10], *ab3=(const float*)d_in[11];  // 128x256
  const float* lw1=(const float*)d_in[12], *lb1=(const float*)d_in[13];  // 3x32
  const float* lw2=(const float*)d_in[14], *lb2=(const float*)d_in[15];  // 32x64
  const float* lw3=(const float*)d_in[16], *lb3=(const float*)d_in[17];  // 64x128
  const float* pw =(const float*)d_in[18], *pb =(const float*)d_in[19];  // 128x256
  const float* qw =(const float*)d_in[20], *qb =(const float*)d_in[21];
  const float* kw =(const float*)d_in[22], *kb =(const float*)d_in[23];
  const float* vw =(const float*)d_in[24], *vb =(const float*)d_in[25];
  const float* ow =(const float*)d_in[26], *ob =(const float*)d_in[27];
  const float* ln1g=(const float*)d_in[28], *ln1b=(const float*)d_in[29];
  const float* ln2g=(const float*)d_in[30], *ln2b=(const float*)d_in[31];
  const float* f1w=(const float*)d_in[32], *f1b=(const float*)d_in[33];  // 256x1024
  const float* f2w=(const float*)d_in[34], *f2b=(const float*)d_in[35];  // 1024x256
  const float* c1w=(const float*)d_in[36], *c1b=(const float*)d_in[37];  // 256x256
  const float* c2w=(const float*)d_in[38], *c2b=(const float*)d_in[39];  // 256x3
  float* out = (float*)d_out;                        // (B,T,3)

  // ---- workspace carve-out ----
  float* ws = (float*)d_ws;
  size_t off = 0;
  auto alloc = [&](size_t n) { float* p = ws + off; off += (n + 63) & ~(size_t)63; return p; };
  float* lane_feat = alloc((size_t)BB * N_LANES * D_LANE);   // (B,NL,128)
  float* map_proj  = alloc((size_t)BB * N_LANES * DD);       // (B,NL,256)
  float* k_map     = alloc((size_t)BB * N_LANES * DD);
  float* v_map     = alloc((size_t)BB * N_LANES * DD);
  float* mq        = alloc((size_t)BB * DD);
  float* mq1       = alloc((size_t)BB * DD);
  float* ego       = alloc((size_t)BB * 3);
  float* h1        = alloc((size_t)BB * N_AG * 64);
  float* h2        = alloc((size_t)BB * N_AG * 128);
  float* pa        = alloc((size_t)BB * N_AG * DD);
  float* distb     = alloc((size_t)BB * N_AG);
  float* pa_sel    = alloc((size_t)BB * K_NN * DD);
  float* k_sel     = alloc((size_t)BB * K_NN * DD);
  float* v_sel     = alloc((size_t)BB * K_NN * DD);
  float* qbuf      = alloc((size_t)BB * DD);
  float* attn_out  = alloc((size_t)BB * DD);
  float* proj      = alloc((size_t)BB * DD);
  float* ffn_h     = alloc((size_t)BB * 4 * DD);
  float* ffn_o     = alloc((size_t)BB * DD);
  float* act_h     = alloc((size_t)BB * DD);

  auto ggrid = [](int M, int N) { return dim3((unsigned)(N / 64), (unsigned)(M / 128)); };
  const int MROWS = BB * N_AG;       // 8192
  const int MSEL  = BB * K_NN;       // 4096

  // ================= hoisted, step-invariant =================
  lane_encoder<<<BB * N_LANES, 128, 0, stream>>>(map_lanes, lanes_mask,
      lw1, lb1, lw2, lb2, lw3, lb3, lane_feat);
  // map_proj = lane_feat @ lane_proj
  wmma_gemm<<<ggrid(MROWS, DD), 256, 0, stream>>>(lane_feat, pw, pb, nullptr,
      map_proj, MROWS, DD, D_LANE, 0);
  // hoisted K/V of map tokens (linear => step-invariant)
  wmma_gemm<<<ggrid(MROWS, DD), 256, 0, stream>>>(map_proj, kw, kb, nullptr,
      k_map, MROWS, DD, DD, 0);
  wmma_gemm<<<ggrid(MROWS, DD), 256, 0, stream>>>(map_proj, vw, vb, nullptr,
      v_map, MROWS, DD, DD, 0);
  init_state<<<(BB * DD + 255) / 256, 256, 0, stream>>>(mode_embed, mode_c, mq, ego);

  // ================= 80 sequential steps =================
  for (int t = 0; t < T_FUT; ++t) {
    agent_pre<<<(MROWS + 255) / 256, 256, 0, stream>>>(agents_seq, t, ego,
        aw1, ab1, h1, distb);
    // agent MLP layers 2,3 (WMMA)
    wmma_gemm<<<ggrid(MROWS, 128), 256, 0, stream>>>(h1, aw2, ab2, nullptr,
        h2, MROWS, 128, 64, 1);
    wmma_gemm<<<ggrid(MROWS, DD), 256, 0, stream>>>(h2, aw3, ab3, agents_mask,
        pa, MROWS, DD, 128, 0);
    topk_gather<<<BB, 64, 0, stream>>>(distb, pa, pa_sel);
    // K/V of selected agents
    wmma_gemm<<<ggrid(MSEL, DD), 256, 0, stream>>>(pa_sel, kw, kb, nullptr,
        k_sel, MSEL, DD, DD, 0);
    wmma_gemm<<<ggrid(MSEL, DD), 256, 0, stream>>>(pa_sel, vw, vb, nullptr,
        v_sel, MSEL, DD, DD, 0);
    // q over batch
    wmma_gemm<<<ggrid(BB, DD), 256, 0, stream>>>(mq, qw, qb, nullptr,
        qbuf, BB, DD, DD, 0);
    attention<<<BB, 256, 0, stream>>>(qbuf, k_sel, k_map, v_sel, v_map, attn_out);
    // out projection + residual LN1
    wmma_gemm<<<ggrid(BB, DD), 256, 0, stream>>>(attn_out, ow, ob, nullptr,
        proj, BB, DD, DD, 0);
    add_ln<<<BB, 256, 0, stream>>>(mq, proj, ln1g, ln1b, mq1);
    // FFN (gelu exact) + residual LN2 -> new carry mq
    wmma_gemm<<<ggrid(BB, 4 * DD), 256, 0, stream>>>(mq1, f1w, f1b, nullptr,
        ffn_h, BB, 4 * DD, DD, 2);
    wmma_gemm<<<ggrid(BB, DD), 256, 0, stream>>>(ffn_h, f2w, f2b, nullptr,
        ffn_o, BB, DD, 4 * DD, 0);
    add_ln<<<BB, 256, 0, stream>>>(mq1, ffn_o, ln2g, ln2b, mq);
    // action head
    wmma_gemm<<<ggrid(BB, DD), 256, 0, stream>>>(mq, c1w, c1b, nullptr,
        act_h, BB, DD, DD, 1);
    action_final<<<BB, 32, 0, stream>>>(act_h, c2w, c2b, ego, out, t);
  }
}
